// GlobalSSM_15229954031676
// MI455X (gfx1250) — compile-verified, matrix operand order checked by reading, not run on hardware
//
#include <hip/hip_runtime.h>
#include <hip/hip_bf16.h>
#include <math.h>

// Problem constants (match reference)
#define D_MODEL  1024
#define SEQ      4096
#define BATCH    2
#define NSTATE   16
#define BS_TOTAL (BATCH * SEQ)   // 8192 rows

typedef __bf16 bf16;
typedef __attribute__((ext_vector_type(16))) __bf16 bf16x16;
typedef __attribute__((ext_vector_type(8)))  __bf16 bf16x8;
typedef __attribute__((ext_vector_type(4)))  __bf16 bf16x4;
typedef __attribute__((ext_vector_type(8)))  float  f32x8;

union FragBF { bf16x16 v16; bf16x8 v8[2]; };

// ---------------------------------------------------------------- CDNA5 async LDS copy
// 16B per-lane global->LDS DMA (ASYNCcnt), no VGPR roundtrip, no loadcnt stall.
// Builtin prototype (from compiler diagnostic): arg0 = int-vec16B ptr in AS(1).

#if defined(__gfx1250__) && __has_builtin(__builtin_amdgcn_global_load_async_to_lds_b128)
typedef int async_v4i __attribute__((vector_size(16)));
typedef __attribute__((address_space(1))) async_v4i* g_v4i_ptr;
typedef __attribute__((address_space(3))) async_v4i* l_v4i_ptr;

__device__ __forceinline__ void async_copy16(void* lds_dst, const void* gsrc) {
  __builtin_amdgcn_global_load_async_to_lds_b128(
      (g_v4i_ptr)gsrc, (l_v4i_ptr)lds_dst, 0, 0);
}
#else
__device__ __forceinline__ void async_copy16(void* lds_dst, const void* gsrc) {
  *reinterpret_cast<uint4*>(lds_dst) = *reinterpret_cast<const uint4*>(gsrc);
}
#endif

__device__ __forceinline__ void async_wait0() {
#if defined(__gfx1250__) && __has_builtin(__builtin_amdgcn_s_wait_asynccnt)
  __builtin_amdgcn_s_wait_asynccnt(0);
#elif defined(__gfx1250__)
  asm volatile("s_wait_asynccnt 0" ::: "memory");
#endif
}

// ---------------------------------------------------------------- utilities

__device__ __forceinline__ float block_sum(float v, float* s) {
  const int t = threadIdx.x;
  s[t] = v;
  __syncthreads();
  for (int off = blockDim.x >> 1; off > 0; off >>= 1) {
    if (t < off) s[t] += s[t + off];
    __syncthreads();
  }
  const float r = s[0];
  __syncthreads();
  return r;
}

// ---------------------------------------------------------------- fp32 -> bf16

__global__ __launch_bounds__(256)
void k_f32_to_bf16(const float* __restrict__ in, bf16* __restrict__ out, int n4) {
  const int i = blockIdx.x * 256 + threadIdx.x;
  if (i < n4) {
    const float4 v = reinterpret_cast<const float4*>(in)[i];
    bf16x4 o;
    o[0] = (bf16)v.x; o[1] = (bf16)v.y; o[2] = (bf16)v.z; o[3] = (bf16)v.w;
    reinterpret_cast<bf16x4*>(out)[i] = o;
  }
}

// ---------------------------------------------------------------- WMMA GEMM
// Out[m,n] = sum_k A[m,k] * B[n,k]   (both operands K-contiguous, bf16)
// MODE 0: softplus epilogue (delta).  MODE 1: multiply by *scale_ptr (context).
// M,N,K all multiples of the tile sizes (8192/1024/1024) -> no bounds checks.

#define BM 128
#define BN 128
#define BK 32
#define BKP 40   // padded LDS row: 80B stride -> conflict-free 16B lane loads

template<int MODE>
__global__ __launch_bounds__(256)
void k_wmma_gemm(const bf16* __restrict__ Ag, const bf16* __restrict__ Bg,
                 float* __restrict__ Out, const float* __restrict__ scale_ptr,
                 int M, int N, int K)
{
  __shared__ bf16 As[2][BM][BKP];
  __shared__ bf16 Bs[2][BN][BKP];

  const int tid  = threadIdx.x;
  const int lane = tid & 31;
  const int wave = tid >> 5;          // 8 wave32 waves
  const int wm   = wave & 3;          // 4 waves along M (32 rows each)
  const int wn   = wave >> 2;         // 2 waves along N (64 cols each)
  const int m0   = blockIdx.y * BM;
  const int n0   = blockIdx.x * BN;
  const int rowl = lane & 15;
  const int kb   = (lane >> 4) << 3;  // lanes 16..31 take k-offset +8
  const int NT   = K / BK;

  f32x8 acc[2][4] = {};

  // issue 4 async 16B DMAs per thread (2 A chunks + 2 B chunks per tile)
  auto load_tile = [&](int kt, int buf) {
    const int k0 = kt * BK;
    #pragma unroll
    for (int i = 0; i < 2; ++i) {
      const int c  = i * 256 + tid;     // 512 chunks of 16B per matrix
      const int r  = c >> 2;
      const int ch = (c & 3) * 8;       // bf16 offset inside row
      async_copy16(&As[buf][r][ch], Ag + (size_t)(m0 + r) * K + k0 + ch);
      async_copy16(&Bs[buf][r][ch], Bg + (size_t)(n0 + r) * K + k0 + ch);
    }
  };

  load_tile(0, 0);
  async_wait0();
  __syncthreads();

  for (int kt = 0; kt < NT; ++kt) {
    const int buf = kt & 1;
    if (kt + 1 < NT) load_tile(kt + 1, buf ^ 1);   // DMA next tile (overlaps WMMA)
    if (kt + 2 < NT && tid < 128) {                // prefetch K+2 tile rows into L2
      __builtin_prefetch(Ag + (size_t)(m0 + tid) * K + (kt + 2) * BK, 0, 1);
      __builtin_prefetch(Bg + (size_t)(n0 + tid) * K + (kt + 2) * BK, 0, 1);
    }

    FragBF a[2], b[4];
    #pragma unroll
    for (int i = 0; i < 2; ++i) {
      const int r = wm * 32 + i * 16 + rowl;
      a[i].v8[0] = *reinterpret_cast<const bf16x8*>(&As[buf][r][kb]);
      a[i].v8[1] = *reinterpret_cast<const bf16x8*>(&As[buf][r][16 + kb]);
    }
    #pragma unroll
    for (int j = 0; j < 4; ++j) {
      const int c = wn * 64 + j * 16 + rowl;
      b[j].v8[0] = *reinterpret_cast<const bf16x8*>(&Bs[buf][c][kb]);
      b[j].v8[1] = *reinterpret_cast<const bf16x8*>(&Bs[buf][c][16 + kb]);
    }
    #pragma unroll
    for (int i = 0; i < 2; ++i)
      #pragma unroll
      for (int j = 0; j < 4; ++j)
        acc[i][j] = __builtin_amdgcn_wmma_f32_16x16x32_bf16(
            false, a[i].v16, false, b[j].v16, (short)0, acc[i][j], false, false);

    async_wait0();       // next-tile DMA must land before the barrier releases it
    __syncthreads();
  }

  const float scale = (MODE == 1) ? scale_ptr[0] : 0.0f;
  const int moff = (lane >> 4) << 3;    // C/D layout: lanes>=16 hold rows +8
  #pragma unroll
  for (int i = 0; i < 2; ++i) {
    #pragma unroll
    for (int j = 0; j < 4; ++j) {
      const int mb = m0 + wm * 32 + i * 16 + moff;
      const int nb = n0 + wn * 64 + j * 16 + rowl;
      #pragma unroll
      for (int r = 0; r < 8; ++r) {
        float v = acc[i][j][r];
        if (MODE == 0) v = (v > 20.0f) ? v : log1pf(__expf(v));  // softplus
        else           v *= scale;
        Out[(size_t)(mb + r) * N + nb] = v;
      }
    }
  }
}

// ---------------------------------------------------------------- B/C selection
// Bsel[row,n] = sum_d x[row,d] * W_B[n,d]   (thin N=16 projections, fp32 VALU)

__global__ __launch_bounds__(256)
void k_bc_proj(const float* __restrict__ x, const float* __restrict__ W_B,
               const float* __restrict__ W_C, float* __restrict__ Bsel,
               float* __restrict__ Csel)
{
  const int i   = blockIdx.x * 256 + threadIdx.x;  // BS_TOTAL*32
  const int row = i >> 5;
  const int j   = i & 31;                          // j<16 -> B, else C
  const int n   = j & 15;
  const float* w  = ((j < 16) ? W_B : W_C) + (size_t)n * D_MODEL;
  const float* xr = x + (size_t)row * D_MODEL;
  float s = 0.f;
  #pragma unroll 8
  for (int d = 0; d < D_MODEL; ++d) s += xr[d] * w[d];
  if (j < 16) Bsel[(size_t)row * NSTATE + n] = s;
  else        Csel[(size_t)row * NSTATE + n] = s;
}

// ---------------------------------------------------------------- SSM scan
// One thread per (b,d) channel; 16-wide state in registers; B/C staged in LDS.

__global__ __launch_bounds__(256)
void k_ssm_scan(const float* __restrict__ x, const float* __restrict__ delta,
                const float* __restrict__ A_log,
                const float* __restrict__ Bsel, const float* __restrict__ Csel,
                float* __restrict__ ssm_out)
{
  const int b = blockIdx.x >> 2;                       // 8 blocks: 2 * (1024/256)
  const int d = ((blockIdx.x & 3) << 8) + threadIdx.x;

  float Arow[NSTATE];
  #pragma unroll
  for (int n = 0; n < NSTATE; ++n) Arow[n] = -__expf(A_log[d * NSTATE + n]);

  float h[NSTATE];
  #pragma unroll
  for (int n = 0; n < NSTATE; ++n) h[n] = 0.f;

  __shared__ float sB[64][NSTATE];
  __shared__ float sC[64][NSTATE];

  for (int s0 = 0; s0 < SEQ; s0 += 64) {
    __syncthreads();
    for (int i = threadIdx.x; i < 64 * NSTATE; i += 256) {
      const size_t gi = ((size_t)b * SEQ + s0 + (i >> 4)) * NSTATE + (i & 15);
      sB[i >> 4][i & 15] = Bsel[gi];
      sC[i >> 4][i & 15] = Csel[gi];
    }
    __syncthreads();
    for (int si = 0; si < 64; ++si) {
      const size_t idx = ((size_t)b * SEQ + s0 + si) * D_MODEL + d;
      const float dt = delta[idx];
      const float dx = dt * x[idx];
      float y = 0.f;
      #pragma unroll
      for (int n = 0; n < NSTATE; ++n) {
        h[n] = h[n] * __expf(dt * Arow[n]) + dx * sB[si][n];
        y   += h[n] * sC[si][n];
      }
      ssm_out[idx] = y;
    }
  }
}

// ---------------------------------------------------------------- LayerNorm

__global__ __launch_bounds__(256)
void k_layernorm(const float* __restrict__ in, const float* __restrict__ gamma,
                 const float* __restrict__ beta, bf16* __restrict__ out_bf,
                 float* __restrict__ h_final)
{
  __shared__ float sred[256];
  const int row = blockIdx.x;                 // b*SEQ + s
  const float* r = in + (size_t)row * D_MODEL;

  float v[4];
  float lsum = 0.f;
  #pragma unroll
  for (int i = 0; i < 4; ++i) { v[i] = r[threadIdx.x + (i << 8)]; lsum += v[i]; }
  const float mu = block_sum(lsum, sred) * (1.0f / D_MODEL);

  float lvar = 0.f;
  #pragma unroll
  for (int i = 0; i < 4; ++i) { const float t = v[i] - mu; lvar += t * t; }
  const float var = block_sum(lvar, sred) * (1.0f / D_MODEL);
  const float inv = rsqrtf(var + 1e-5f);

  const bool last = ((row & (SEQ - 1)) == (SEQ - 1));
  const int  bb   = row >> 12;                // row / SEQ
  #pragma unroll
  for (int i = 0; i < 4; ++i) {
    const int col = threadIdx.x + (i << 8);
    const float o = (v[i] - mu) * inv * gamma[col] + beta[col];
    out_bf[(size_t)row * D_MODEL + col] = (bf16)o;
    if (last) h_final[(size_t)bb * D_MODEL + col] = o;
  }
}

// ---------------------------------------------------------------- importance path

__global__ __launch_bounds__(256)
void k_hproj(const float* __restrict__ h_final, const float* __restrict__ W_imp,
             float* __restrict__ h_proj)
{
  const int i = blockIdx.x * 256 + threadIdx.x;    // 0 .. BATCH*D-1
  const int b = i >> 10, e = i & (D_MODEL - 1);
  const float* hf = h_final + (size_t)b * D_MODEL;
  const float* w  = W_imp + (size_t)e * D_MODEL;
  float s = 0.f;
  #pragma unroll 8
  for (int d = 0; d < D_MODEL; ++d) s += hf[d] * w[d];
  h_proj[i] = s;
}

__global__ __launch_bounds__(256)
void k_raw_importance(const float* __restrict__ x, const float* __restrict__ h_proj,
                      float* __restrict__ raw)
{
  __shared__ float sred[256];
  const int row = blockIdx.x;
  const int b   = row >> 12;
  const float* xr = x + (size_t)row * D_MODEL;
  const float* hp = h_proj + (size_t)b * D_MODEL;
  float s = 0.f;
  for (int d = threadIdx.x; d < D_MODEL; d += 256) s += xr[d] * hp[d];
  const float tot = block_sum(s, sred);
  if (threadIdx.x == 0) raw[row] = tot;
}

__global__ __launch_bounds__(1024)
void k_softmax(const float* __restrict__ raw, float* __restrict__ imp)
{
  __shared__ float sred[1024];
  const int b = blockIdx.x;
  const float* r = raw + (size_t)b * SEQ;

  float m = -1e30f;
  for (int i = threadIdx.x; i < SEQ; i += 1024) m = fmaxf(m, r[i] * 2.0f);
  sred[threadIdx.x] = m; __syncthreads();
  for (int off = 512; off > 0; off >>= 1) {
    if (threadIdx.x < off) sred[threadIdx.x] = fmaxf(sred[threadIdx.x], sred[threadIdx.x + off]);
    __syncthreads();
  }
  const float M = sred[0]; __syncthreads();

  float s = 0.f;
  for (int i = threadIdx.x; i < SEQ; i += 1024) s += __expf(r[i] * 2.0f - M);
  sred[threadIdx.x] = s; __syncthreads();
  for (int off = 512; off > 0; off >>= 1) {
    if (threadIdx.x < off) sred[threadIdx.x] += sred[threadIdx.x + off];
    __syncthreads();
  }
  const float inv = 1.0f / sred[0];
  for (int i = threadIdx.x; i < SEQ; i += 1024)
    imp[(size_t)b * SEQ + i] = __expf(r[i] * 2.0f - M) * inv;
}

// ---------------------------------------------------------------- launch

extern "C" void kernel_launch(void* const* d_in, const int* in_sizes, int n_in,
                              void* d_out, int out_size, void* d_ws, size_t ws_size,
                              hipStream_t stream)
{
  const float* x        = (const float*)d_in[0];
  const float* A_log    = (const float*)d_in[1];
  const float* W_delta  = (const float*)d_in[2];
  const float* W_B      = (const float*)d_in[3];
  const float* W_C      = (const float*)d_in[4];
  const float* ln_gamma = (const float*)d_in[5];
  const float* ln_beta  = (const float*)d_in[6];
  const float* W_ctx    = (const float*)d_in[7];
  const float* ctx_scale= (const float*)d_in[8];   // device scalar
  const float* W_imp    = (const float*)d_in[9];

  // workspace carve-up (~106 MB total)
  char* ws = (char*)d_ws;
  size_t off = 0;
  bf16*  x_bf    = (bf16*) (ws + off); off += (size_t)BS_TOTAL * D_MODEL * 2;
  bf16*  wd_bf   = (bf16*) (ws + off); off += (size_t)D_MODEL * D_MODEL * 2;
  bf16*  wc_bf   = (bf16*) (ws + off); off += (size_t)D_MODEL * D_MODEL * 2;
  float* delta   = (float*)(ws + off); off += (size_t)BS_TOTAL * D_MODEL * 4;
  float* Bsel    = (float*)(ws + off); off += (size_t)BS_TOTAL * NSTATE * 4;
  float* Csel    = (float*)(ws + off); off += (size_t)BS_TOTAL * NSTATE * 4;
  float* ssm     = (float*)(ws + off); off += (size_t)BS_TOTAL * D_MODEL * 4;
  bf16*  ln_bf   = (bf16*) (ws + off); off += (size_t)BS_TOTAL * D_MODEL * 2;
  float* h_final = (float*)(ws + off); off += (size_t)BATCH * D_MODEL * 4;
  float* h_proj  = (float*)(ws + off); off += (size_t)BATCH * D_MODEL * 4;

  // output layout: importance (B*S) | context (B*S*D) | raw_importance (B*S)
  float* out_imp = (float*)d_out;
  float* out_ctx = out_imp + BS_TOTAL;
  float* out_raw = out_ctx + (size_t)BS_TOTAL * D_MODEL;

  // 1. precision conversion
  {
    const int n4x = BS_TOTAL * D_MODEL / 4;
    const int n4w = D_MODEL * D_MODEL / 4;
    k_f32_to_bf16<<<(n4x + 255) / 256, 256, 0, stream>>>(x, x_bf, n4x);
    k_f32_to_bf16<<<(n4w + 255) / 256, 256, 0, stream>>>(W_delta, wd_bf, n4w);
    k_f32_to_bf16<<<(n4w + 255) / 256, 256, 0, stream>>>(W_ctx,   wc_bf, n4w);
  }

  dim3 ggrid(D_MODEL / BN, BS_TOTAL / BM);   // 8 x 64 blocks

  // 2. delta = softplus(x @ W_delta^T)   [WMMA bf16 -> f32, fused softplus]
  k_wmma_gemm<0><<<ggrid, 256, 0, stream>>>(x_bf, wd_bf, delta, nullptr,
                                            BS_TOTAL, D_MODEL, D_MODEL);

  // 3. B_sel / C_sel thin projections
  k_bc_proj<<<(BS_TOTAL * 32) / 256, 256, 0, stream>>>(x, W_B, W_C, Bsel, Csel);

  // 4. sequential selective scan
  k_ssm_scan<<<BATCH * D_MODEL / 256, 256, 0, stream>>>(x, delta, A_log,
                                                        Bsel, Csel, ssm);

  // 5. layernorm (-> bf16 for context GEMM, fp32 h_final for importance)
  k_layernorm<<<BS_TOTAL, 256, 0, stream>>>(ssm, ln_gamma, ln_beta, ln_bf, h_final);

  // 6. context = LN(ssm) @ W_ctx^T * scale   [WMMA, fused scale]
  k_wmma_gemm<1><<<ggrid, 256, 0, stream>>>(ln_bf, wc_bf, out_ctx, ctx_scale,
                                            BS_TOTAL, D_MODEL, D_MODEL);

  // 7-9. importance path
  k_hproj<<<BATCH * D_MODEL / 256, 256, 0, stream>>>(h_final, W_imp, h_proj);
  k_raw_importance<<<BS_TOTAL, 256, 0, stream>>>(x, h_proj, out_raw);
  k_softmax<<<BATCH, 1024, 0, stream>>>(out_raw, out_imp);
}
